// DDH_17059610100509
// MI455X (gfx1250) — matrix operand order repaired and checked by reading, not compiled, not run on hardware
//
#include <hip/hip_runtime.h>
#include <hip/hip_bf16.h>

#define BATCH 4096
#define KP    864          // FC: 860 padded to multiple of 32
#define NFC   192
#define NHASH 48
#define SPLIT 4
#define NCLS  10

typedef __attribute__((ext_vector_type(16))) _Float16 v16h;
typedef __attribute__((ext_vector_type(8)))  float    v8f;

// ---------------------------------------------------------------------------
// sum/sumsq -> per-channel affine (a = g*rsqrt(var+eps), b = beta - mean*a)
// ---------------------------------------------------------------------------
__global__ void bnparam(const float* __restrict__ gsum, const float* __restrict__ gsumsq,
                        const float* __restrict__ g, const float* __restrict__ be,
                        float* __restrict__ a, float* __restrict__ bb,
                        int nc, float invcount) {
    int c = blockIdx.x * blockDim.x + threadIdx.x;
    if (c < nc) {
        float m = gsum[c] * invcount;
        float v = gsumsq[c] * invcount - m * m;
        float s = rsqrtf(v + 1e-5f) * g[c];
        a[c]  = s;
        bb[c] = be[c] - m * s;
    }
}

// ---------------------------------------------------------------------------
// Pack conv weights (COUT, CIN*KH*KW) into zero-padded f16 (NP x KPAD), B^T rows
// ---------------------------------------------------------------------------
__global__ void fill_wb(const float* __restrict__ w, _Float16* __restrict__ wbh,
                        int cout, int kreal, int kpad, int np) {
    int idx = blockIdx.x * blockDim.x + threadIdx.x;
    int total = np * kpad;
    if (idx >= total) return;
    int k = idx % kpad, n = idx / kpad;
    wbh[idx] = (n < cout && k < kreal) ? (_Float16)w[n * kreal + k] : (_Float16)0.f;
}

// ---------------------------------------------------------------------------
// Implicit-GEMM conv, stats pass: one wave = 16(M=ox) x 16(N=oc) tile of conv
// output via v_wmma_f32_16x16x32_f16; per-channel sum/sumsq via LDS + atomics.
// A gathered from x (f32 -> f16) with LDS offset table; K / ox padding by
// branchless clamps so EXEC is all-ones at the WMMAs.
// ---------------------------------------------------------------------------
template<int CIN,int COUT,int KH,int IN,int OUT,int KSTEPS,int NT,int OXT>
__global__ void conv_gemm_stats(const float* __restrict__ x, const _Float16* __restrict__ wbh,
                                const float* __restrict__ bias,
                                float* __restrict__ gsum, float* __restrict__ gsumsq) {
    constexpr int KREAL = CIN * KH * KH;
    constexpr int KPAD  = KSTEPS * 32;
    constexpr int NP    = NT * 16;
    __shared__ int   F[KREAL];
    __shared__ float ls[NP], lq[NP];
    int t = threadIdx.x;
    for (int i = t; i < KREAL; i += blockDim.x) {
        int c = i / (KH * KH), rem = i % (KH * KH);
        F[i] = c * IN * IN + (rem / KH) * IN + (rem % KH);
    }
    for (int i = t; i < NP; i += blockDim.x) { ls[i] = 0.f; lq[i] = 0.f; }
    __syncthreads();

    int wave  = (blockIdx.x * blockDim.x + t) >> 5;
    int lane  = t & 31;
    int nTile = wave % NT;  int r1 = wave / NT;
    int oxT   = r1 % OXT;   int r2 = r1 / OXT;
    int oy    = r2 % OUT;   int b  = r2 / OUT;

    int hi   = lane >> 4;        // K-half select for A/B frags, M-half for C
    int colA = lane & 15;        // A: M row ; B/C: N column
    int ox   = oxT * 16 + colA;
    int oxc  = (ox < OUT) ? ox : (OUT - 1);
    const float* xb = x + (long)b * CIN * IN * IN + oy * IN + oxc;
    int n = nTile * 16 + colA;
    const _Float16* wrow = wbh + (long)n * KPAD;

    v8f acc = {};
    #pragma unroll
    for (int ks = 0; ks < KSTEPS; ++ks) {
        int k0 = ks * 32;
        v16h A, Bm;
        #pragma unroll
        for (int h = 0; h < 16; ++h) {
            int vg = h >> 1, sb = h & 1;
            int kk = k0 + ((vg >= 4) ? 16 : 0) + (hi ? 8 : 0) + 2 * (vg & 3) + sb;
            int ki = (kk < KREAL) ? kk : 0;
            float v = xb[F[ki]];
            A[h] = (_Float16)((kk < KREAL) ? v : 0.f);
        }
        #pragma unroll
        for (int h = 0; h < 16; ++h)
            Bm[h] = wrow[k0 + ((h >= 8) ? 16 : 0) + (hi ? 8 : 0) + (h & 7)];
        acc = __builtin_amdgcn_wmma_f32_16x16x32_f16(false, A, false, Bm,
                                                     (short)0, acc, false, false);
    }
    // C layout: VGPR r holds M-row r + hi*8, column = lane&15
    float bi = bias[(n < COUT) ? n : 0];
    float s = 0.f, q = 0.f;
    #pragma unroll
    for (int r = 0; r < 8; ++r) {
        int oxr = oxT * 16 + r + hi * 8;
        float v = acc[r] + bi;
        v = (oxr < OUT) ? v : 0.f;
        s += v;
        q += v * v;
    }
    if (n < COUT) {
        atomicAdd(&ls[n], s);
        atomicAdd(&lq[n], q);
    }
    __syncthreads();
    for (int i = t; i < COUT; i += blockDim.x) {
        atomicAdd(&gsum[i],   ls[i]);
        atomicAdd(&gsumsq[i], lq[i]);
    }
}

// ---------------------------------------------------------------------------
// Implicit-GEMM conv, apply pass: one wave computes rows oy=2py,2py+1 (two
// accumulators, shared B frag), then BN affine + ReLU + 2x2 maxpool entirely
// in registers (horizontal partners are adjacent C elements in-lane, vertical
// partner is in the second accumulator). Writes pooled (B,COUT,PO,PO).
// ---------------------------------------------------------------------------
template<int CIN,int COUT,int KH,int IN,int OUT,int KSTEPS,int NT,int OXT>
__global__ void conv_gemm_pool(const float* __restrict__ x, const _Float16* __restrict__ wbh,
                               const float* __restrict__ bias,
                               const float* __restrict__ a, const float* __restrict__ bb,
                               float* __restrict__ out) {
    constexpr int KREAL = CIN * KH * KH;
    constexpr int KPAD  = KSTEPS * 32;
    constexpr int PO    = OUT / 2;
    __shared__ int F[KREAL];
    int t = threadIdx.x;
    for (int i = t; i < KREAL; i += blockDim.x) {
        int c = i / (KH * KH), rem = i % (KH * KH);
        F[i] = c * IN * IN + (rem / KH) * IN + (rem % KH);
    }
    __syncthreads();

    int wave  = (blockIdx.x * blockDim.x + t) >> 5;
    int lane  = t & 31;
    int nTile = wave % NT;  int r1 = wave / NT;
    int oxT   = r1 % OXT;   int r2 = r1 / OXT;
    int py    = r2 % PO;    int b  = r2 / PO;
    int oy0   = 2 * py;

    int hi   = lane >> 4;
    int colA = lane & 15;
    int ox   = oxT * 16 + colA;
    int oxc  = (ox < OUT) ? ox : (OUT - 1);
    const float* xb0 = x + (long)b * CIN * IN * IN + oy0 * IN + oxc;
    const float* xb1 = xb0 + IN;
    int n = nTile * 16 + colA;
    const _Float16* wrow = wbh + (long)n * KPAD;

    v8f acc0 = {}, acc1 = {};
    #pragma unroll
    for (int ks = 0; ks < KSTEPS; ++ks) {
        int k0 = ks * 32;
        v16h A0, A1, Bm;
        #pragma unroll
        for (int h = 0; h < 16; ++h) {
            int vg = h >> 1, sb = h & 1;
            int kk = k0 + ((vg >= 4) ? 16 : 0) + (hi ? 8 : 0) + 2 * (vg & 3) + sb;
            int ki = (kk < KREAL) ? kk : 0;
            int f  = F[ki];
            bool ok = (kk < KREAL);
            float v0 = xb0[f], v1 = xb1[f];
            A0[h] = (_Float16)(ok ? v0 : 0.f);
            A1[h] = (_Float16)(ok ? v1 : 0.f);
        }
        #pragma unroll
        for (int h = 0; h < 16; ++h)
            Bm[h] = wrow[k0 + ((h >= 8) ? 16 : 0) + (hi ? 8 : 0) + (h & 7)];
        acc0 = __builtin_amdgcn_wmma_f32_16x16x32_f16(false, A0, false, Bm,
                                                      (short)0, acc0, false, false);
        acc1 = __builtin_amdgcn_wmma_f32_16x16x32_f16(false, A1, false, Bm,
                                                      (short)0, acc1, false, false);
    }
    int oc = n;
    if (oc < COUT) {
        float bi = bias[oc];
        float sc = a[oc], sh = bb[oc];
        #pragma unroll
        for (int p = 0; p < 4; ++p) {
            int px = oxT * 8 + p + hi * 4;   // rows 2p,2p+1 <-> ox pair (2px,2px+1)
            if (px < PO) {
                float v00 = fmaxf((acc0[2 * p]     + bi) * sc + sh, 0.f);
                float v01 = fmaxf((acc0[2 * p + 1] + bi) * sc + sh, 0.f);
                float v10 = fmaxf((acc1[2 * p]     + bi) * sc + sh, 0.f);
                float v11 = fmaxf((acc1[2 * p + 1] + bi) * sc + sh, 0.f);
                float m = fmaxf(fmaxf(v00, v01), fmaxf(v10, v11));
                out[((long)(b * COUT + oc) * PO + py) * PO + px] = m;
            }
        }
    }
}

// ---------------------------------------------------------------------------
// Locally-connected 60->80, unshared 2x2 kernels, plus channel stats (scalar)
// ---------------------------------------------------------------------------
__global__ void local_stats(const float* __restrict__ h3, const float* __restrict__ wl,
                            const float* __restrict__ bl,
                            float* __restrict__ y4,
                            float* __restrict__ gsum, float* __restrict__ gsumsq) {
    __shared__ float ls[80], lq[80];
    int t = threadIdx.x;
    for (int i = t; i < 80; i += blockDim.x) { ls[i] = 0.f; lq[i] = 0.f; }
    __syncthreads();

    int idx = blockIdx.x * blockDim.x + t;
    const int total = BATCH * 80 * 4;
    if (idx < total) {
        int j = idx % 2;  int r = idx / 2;
        int i = r % 2;    r /= 2;
        int o = r % 80;   int b = r / 80;
        float acc = bl[(o * 2 + i) * 2 + j];
        const float* hb = h3 + (long)b * 60 * 9;
        const float* wp = wl + (long)((i * 2 + j) * 80 + o) * 60 * 4;
        #pragma unroll
        for (int c = 0; c < 60; ++c)
            #pragma unroll
            for (int k = 0; k < 2; ++k)
                #pragma unroll
                for (int l = 0; l < 2; ++l)
                    acc += hb[c * 9 + (i + k) * 3 + (j + l)] * wp[(c * 2 + k) * 2 + l];
        y4[(long)b * 320 + o * 4 + i * 2 + j] = acc;
        atomicAdd(&ls[o], acc);
        atomicAdd(&lq[o], acc * acc);
    }
    __syncthreads();
    for (int i = t; i < 80; i += blockDim.x) {
        atomicAdd(&gsum[i],   ls[i]);
        atomicAdd(&gsumsq[i], lq[i]);
    }
}

// ---------------------------------------------------------------------------
// Build padded f16 activation matrix zh (BATCH x KP): [h3 flat | BN+ReLU(y4) | 0]
// ---------------------------------------------------------------------------
__global__ void fill_zh(const float* __restrict__ h3, const float* __restrict__ y4,
                        const float* __restrict__ a4, const float* __restrict__ b4,
                        _Float16* __restrict__ zh) {
    int idx = blockIdx.x * blockDim.x + threadIdx.x;
    const int total = BATCH * KP;
    if (idx >= total) return;
    int k = idx % KP, b = idx / KP;
    float v = 0.f;
    if (k < 540) {
        v = h3[(long)b * 540 + k];
    } else if (k < 860) {
        int k2 = k - 540;
        int o = k2 >> 2;
        float y = y4[(long)b * 320 + k2];
        v = fmaxf(y * a4[o] + b4[o], 0.f);
    }
    zh[(long)b * KP + k] = (_Float16)v;
}

__global__ void fill_wh(const float* __restrict__ wfc, _Float16* __restrict__ wh) {
    int idx = blockIdx.x * blockDim.x + threadIdx.x;
    const int total = NFC * KP;
    if (idx >= total) return;
    int k = idx % KP, n = idx / KP;
    wh[(long)n * KP + k] = (k < 860) ? (_Float16)wfc[(long)n * 860 + k] : (_Float16)0.f;
}

// ---------------------------------------------------------------------------
// FC GEMM via v_wmma_f32_16x16x32_f16: zfc[4096x192] = zh * wh^T + bfc
// ---------------------------------------------------------------------------
__global__ void fc_wmma(const _Float16* __restrict__ zh, const _Float16* __restrict__ wh,
                        const float* __restrict__ bfc, float* __restrict__ zfc) {
    const int NT = NFC / 16;
    int wave  = (blockIdx.x * blockDim.x + threadIdx.x) >> 5;
    int lane  = threadIdx.x & 31;
    int mTile = wave / NT;
    int nTile = wave % NT;

    int hi   = lane >> 4;
    int rowA = lane & 15;

    const _Float16* za = zh + (long)(mTile * 16 + rowA) * KP;
    const _Float16* wb = wh + (long)(nTile * 16 + rowA) * KP;

    v8f c = {};
    for (int k0 = 0; k0 < KP; k0 += 32) {
        __builtin_prefetch(za + k0 + 32, 0, 0);
        v16h A, Bm;
        #pragma unroll
        for (int h = 0; h < 16; ++h) {
            int vg = h >> 1, sb = h & 1;
            int kk = ((vg >= 4) ? 16 : 0) + (hi ? 8 : 0) + 2 * (vg & 3) + sb;
            A[h] = za[k0 + kk];
        }
        #pragma unroll
        for (int h = 0; h < 16; ++h)
            Bm[h] = wb[k0 + ((h >= 8) ? 16 : 0) + (hi ? 8 : 0) + (h & 7)];
        c = __builtin_amdgcn_wmma_f32_16x16x32_f16(false, A, false, Bm,
                                                   (short)0, c, false, false);
    }
    int n = nTile * 16 + rowA;
    float bn = bfc[n];
    #pragma unroll
    for (int r = 0; r < 8; ++r) {
        int m = mTile * 16 + r + (hi ? 8 : 0);
        zfc[(long)m * NFC + n] = c[r] + bn;
    }
}

// ---------------------------------------------------------------------------
// Per-feature column stats over the batch (one block per feature)
// ---------------------------------------------------------------------------
__global__ void colstats(const float* __restrict__ z, float* __restrict__ gsum,
                         float* __restrict__ gsumsq, int ncols) {
    __shared__ float ls[256], lq[256];
    int f = blockIdx.x;
    float s = 0.f, q = 0.f;
    for (int r = threadIdx.x; r < BATCH; r += blockDim.x) {
        float v = z[(long)r * ncols + f];
        s += v; q += v * v;
    }
    ls[threadIdx.x] = s; lq[threadIdx.x] = q;
    __syncthreads();
    for (int st = 128; st > 0; st >>= 1) {
        if (threadIdx.x < st) {
            ls[threadIdx.x] += ls[threadIdx.x + st];
            lq[threadIdx.x] += lq[threadIdx.x + st];
        }
        __syncthreads();
    }
    if (threadIdx.x == 0) { gsum[f] = ls[0]; gsumsq[f] = lq[0]; }
}

// ---------------------------------------------------------------------------
// BN5 affine + ReLU + per-hash SPLIT->1 fuse heads
// ---------------------------------------------------------------------------
__global__ void fuse_heads(const float* __restrict__ zfc,
                           const float* __restrict__ a5, const float* __restrict__ b5,
                           const float* __restrict__ wfuse, const float* __restrict__ bfuse,
                           float* __restrict__ fused) {
    int idx = blockIdx.x * blockDim.x + threadIdx.x;
    const int total = BATCH * NHASH;
    if (idx >= total) return;
    int h = idx % NHASH, b = idx / NHASH;
    float acc = bfuse[h];
    #pragma unroll
    for (int s = 0; s < SPLIT; ++s) {
        int f = h * SPLIT + s;
        float v = zfc[(long)b * NFC + f];
        v = fmaxf(v * a5[f] + b5[f], 0.f);
        acc += v * wfuse[h * SPLIT + s];
    }
    fused[(long)b * NHASH + h] = acc;
}

// ---------------------------------------------------------------------------
// BN6 affine + tanh -> codes; codes @ wlast.T + blast -> softmax -> probs
// ---------------------------------------------------------------------------
__global__ void final_head(const float* __restrict__ fused,
                           const float* __restrict__ a6, const float* __restrict__ b6,
                           const float* __restrict__ wlast, const float* __restrict__ blast,
                           float* __restrict__ out_codes, float* __restrict__ out_probs) {
    int b = blockIdx.x * blockDim.x + threadIdx.x;
    if (b >= BATCH) return;
    float codes[NHASH];
    #pragma unroll
    for (int h = 0; h < NHASH; ++h) {
        float v = tanhf(fused[(long)b * NHASH + h] * a6[h] + b6[h]);
        codes[h] = v;
        out_codes[(long)b * NHASH + h] = v;
    }
    float logits[NCLS];
    float mx = -1e30f;
    #pragma unroll
    for (int c = 0; c < NCLS; ++c) {
        float l = blast[c];
        #pragma unroll
        for (int h = 0; h < NHASH; ++h) l += codes[h] * wlast[c * NHASH + h];
        logits[c] = l;
        mx = fmaxf(mx, l);
    }
    float se = 0.f;
    #pragma unroll
    for (int c = 0; c < NCLS; ++c) { logits[c] = __expf(logits[c] - mx); se += logits[c]; }
    float inv = 1.f / se;
    #pragma unroll
    for (int c = 0; c < NCLS; ++c) out_probs[(long)b * NCLS + c] = logits[c] * inv;
}

// ---------------------------------------------------------------------------
// Host-side orchestration
// ---------------------------------------------------------------------------
extern "C" void kernel_launch(void* const* d_in, const int* in_sizes, int n_in,
                              void* d_out, int out_size, void* d_ws, size_t ws_size,
                              hipStream_t stream) {
    const float* x     = (const float*)d_in[0];
    const float* w1    = (const float*)d_in[1];
    const float* b1    = (const float*)d_in[2];
    const float* g1    = (const float*)d_in[3];
    const float* be1   = (const float*)d_in[4];
    const float* w2    = (const float*)d_in[5];
    const float* b2    = (const float*)d_in[6];
    const float* g2    = (const float*)d_in[7];
    const float* be2   = (const float*)d_in[8];
    const float* w3    = (const float*)d_in[9];
    const float* b3    = (const float*)d_in[10];
    const float* g3    = (const float*)d_in[11];
    const float* be3   = (const float*)d_in[12];
    const float* wl    = (const float*)d_in[13];
    const float* bl    = (const float*)d_in[14];
    const float* g4    = (const float*)d_in[15];
    const float* be4   = (const float*)d_in[16];
    const float* wfc   = (const float*)d_in[17];
    const float* bfc   = (const float*)d_in[18];
    const float* g5    = (const float*)d_in[19];
    const float* be5   = (const float*)d_in[20];
    const float* wfuse = (const float*)d_in[21];
    const float* bfuse = (const float*)d_in[22];
    const float* g6    = (const float*)d_in[23];
    const float* be6   = (const float*)d_in[24];
    const float* wlast = (const float*)d_in[25];
    const float* blast = (const float*)d_in[26];

    char* ws = (char*)d_ws;
    float*    h1    = (float*)(ws + 0);                // (B,20,15,15) = 73,728,000 B
    float*    h2    = (float*)(ws + 73728000);         // (B,40,7,7)   = 32,112,640 B
    float*    h3    = (float*)(ws + 105840640);        // (B,60,3,3)   =  8,847,360 B
    float*    y4    = (float*)(ws + 114688000);        // (B,80,2,2)   =  5,242,880 B
    float*    zfc   = (float*)(ws + 119930880);        // (B,192)      =  3,145,728 B
    float*    fused = (float*)(ws + 123076608);        // (B,48)       =    786,432 B
    _Float16* zh    = (_Float16*)(ws + 123863040);     // (B,864) f16  =  7,077,888 B
    _Float16* wh    = (_Float16*)(ws + 130940928);     // (192,864)f16 =    331,776 B
    float*    ssum  = (float*)(ws + 131272704);        // 192 f32
    float*    ssq   = (float*)(ws + 131273472);        // 192 f32
    float*    saf   = (float*)(ws + 131274240);        // 192 f32
    float*    sbf   = (float*)(ws + 131275008);        // 192 f32
    _Float16* wb1h  = (_Float16*)(ws + 131275776);     // 32x32  f16 =  2,048 B
    _Float16* wb2h  = (_Float16*)(ws + 131277824);     // 48x96  f16 =  9,216 B
    _Float16* wb3h  = (_Float16*)(ws + 131287040);     // 64x160 f16 = 20,480 B

    float* out_codes = (float*)d_out;
    float* out_probs = out_codes + (long)BATCH * NHASH;

    const int TB = 256;

    // ---- Pack conv weights to padded f16 (B^T rows)
    fill_wb<<<(32 * 32 + TB - 1) / TB, TB, 0, stream>>>(w1, wb1h, 20, 27, 32, 32);
    fill_wb<<<(48 * 96 + TB - 1) / TB, TB, 0, stream>>>(w2, wb2h, 40, 80, 96, 48);
    fill_wb<<<(64 * 160 + TB - 1) / TB, TB, 0, stream>>>(w3, wb3h, 60, 160, 160, 64);

    // ---- C1: implicit-GEMM conv3x3 3->20 (K 27->32), BN, ReLU, pool -> h1
    hipMemsetAsync(ssum, 0, 2 * 192 * sizeof(float), stream);
    conv_gemm_stats<3,20,3,32,30,1,2,2><<<BATCH * 30 * 2 * 2 / 8, TB, 0, stream>>>(
        x, wb1h, b1, ssum, ssq);
    bnparam<<<1, TB, 0, stream>>>(ssum, ssq, g1, be1, saf, sbf, 20, 1.f / (BATCH * 900.f));
    conv_gemm_pool<3,20,3,32,30,1,2,2><<<BATCH * 15 * 2 * 2 / 8, TB, 0, stream>>>(
        x, wb1h, b1, saf, sbf, h1);

    // ---- C2: implicit-GEMM conv2x2 20->40 (K 80->96) -> h2
    hipMemsetAsync(ssum, 0, 2 * 192 * sizeof(float), stream);
    conv_gemm_stats<20,40,2,15,14,3,3,1><<<BATCH * 14 * 3 / 8, TB, 0, stream>>>(
        h1, wb2h, b2, ssum, ssq);
    bnparam<<<1, TB, 0, stream>>>(ssum, ssq, g2, be2, saf, sbf, 40, 1.f / (BATCH * 196.f));
    conv_gemm_pool<20,40,2,15,14,3,3,1><<<BATCH * 7 * 3 / 8, TB, 0, stream>>>(
        h1, wb2h, b2, saf, sbf, h2);

    // ---- C3: implicit-GEMM conv2x2 40->60 (K = 160 exact) -> h3
    hipMemsetAsync(ssum, 0, 2 * 192 * sizeof(float), stream);
    conv_gemm_stats<40,60,2,7,6,5,4,1><<<BATCH * 6 * 4 / 8, TB, 0, stream>>>(
        h2, wb3h, b3, ssum, ssq);
    bnparam<<<1, TB, 0, stream>>>(ssum, ssq, g3, be3, saf, sbf, 60, 1.f / (BATCH * 36.f));
    conv_gemm_pool<40,60,2,7,6,5,4,1><<<BATCH * 3 * 4 / 8, TB, 0, stream>>>(
        h2, wb3h, b3, saf, sbf, h3);

    // ---- Locally-connected 60->80 (pre-BN) -> y4, with stats
    hipMemsetAsync(ssum, 0, 2 * 192 * sizeof(float), stream);
    local_stats<<<BATCH * 320 / TB, TB, 0, stream>>>(h3, wl, bl, y4, ssum, ssq);
    bnparam<<<1, TB, 0, stream>>>(ssum, ssq, g4, be4, saf, sbf, 80, 1.f / (BATCH * 4.f));

    // ---- Assemble padded f16 FC operands
    fill_zh<<<BATCH * KP / TB, TB, 0, stream>>>(h3, y4, saf, sbf, zh);
    fill_wh<<<NFC * KP / TB, TB, 0, stream>>>(wfc, wh);

    // ---- FC GEMM (4096 x 864 x 192) on WMMA
    {
        int tiles  = (BATCH / 16) * (NFC / 16);
        int blocks = tiles / 8;
        fc_wmma<<<blocks, TB, 0, stream>>>(zh, wh, bfc, zfc);
    }

    // ---- BN5 stats + params, fuse heads
    colstats<<<NFC, TB, 0, stream>>>(zfc, ssum, ssq, NFC);
    bnparam<<<1, TB, 0, stream>>>(ssum, ssq, g5, be5, saf, sbf, NFC, 1.f / (float)BATCH);
    fuse_heads<<<BATCH * NHASH / TB, TB, 0, stream>>>(zfc, saf, sbf, wfuse, bfuse, fused);

    // ---- BN6 stats + params, tanh codes + classifier softmax
    colstats<<<NHASH, TB, 0, stream>>>(fused, ssum, ssq, NHASH);
    bnparam<<<1, TB, 0, stream>>>(ssum, ssq, g6, be6, saf, sbf, NHASH, 1.f / (float)BATCH);
    final_head<<<BATCH / TB, TB, 0, stream>>>(fused, saf, sbf, wlast, blast, out_codes, out_probs);
}
